// SpatialSelfAttentionLayer_63771674411396
// MI455X (gfx1250) — compile-verified, matrix operand order checked by reading, not run on hardware
//
#include <hip/hip_runtime.h>
#include <hip/hip_bf16.h>

typedef __bf16 bf16;
typedef __attribute__((ext_vector_type(8)))  bf16  v8bf;
typedef __attribute__((ext_vector_type(16))) bf16  v16bf;
typedef __attribute__((ext_vector_type(8)))  float v8f;

namespace {
constexpr int BATCH = 1024;
constexpr int L     = 100;
constexpr int LP    = 112;    // L padded to 7 tiles of 16
constexpr int D     = 256;
constexpr int NHEAD = 8;
constexpr int DH    = 32;
constexpr int KPAD  = 128;    // key dim padded to 4 k-tiles of 32

// dynamic LDS layout (bytes)
constexpr int OFF_A = 0;       // bf16 [LP][D]   : phase1 h ; phase2+ attention output O
constexpr int OFF_Q = 57344;   // bf16 [LP][D]   ; phase3: f32 residual [LP][D] (spans Q+K)
constexpr int OFF_K = 114688;  // bf16 [LP][D]
constexpr int OFF_V = 172032;  // bf16 [D][KPAD] (V transposed: [feature][key])
constexpr int OFF_S = 237568;  // f32  [LP][LP]  scores
constexpr int OFF_P = 287744;  // bf16 [LP][KPAD] probs
constexpr int OFF_C = 316416;  // f32  [LP][4]   coords
constexpr int OFF_M = 318208;  // u32  [LP][4]   mask bits
constexpr int OFF_R = 320000;  // f32  [64]      reduction scratch
constexpr int SMEM_BYTES = 320256;

constexpr float LOG2BETA = -0.15200309344504997f; // log2(0.9)
constexpr float LOG2E    = 1.4426950408889634f;
constexpr float SCALE    = 0.17677669529663687f;  // 1/sqrt(32)
constexpr float LN_EPS   = 1e-5f;
}

__device__ __forceinline__ v16bf frag16(const bf16* p0, const bf16* p1) {
  v8bf lo = *(const v8bf*)p0;
  v8bf hi = *(const v8bf*)p1;
  return __builtin_shufflevector(lo, hi, 0,1,2,3,4,5,6,7,8,9,10,11,12,13,14,15);
}

__device__ __forceinline__ v8f wmma_bf16(v16bf a, v16bf b, v8f c) {
  return __builtin_amdgcn_wmma_f32_16x16x32_bf16(false, a, false, b, (short)0, c, false, false);
}

__device__ __forceinline__ float wave_sum(float v) {
#pragma unroll
  for (int o = 16; o > 0; o >>= 1) v += __shfl_xor(v, o, 32);
  return v;
}
__device__ __forceinline__ float wave_max(float v) {
#pragma unroll
  for (int o = 16; o > 0; o >>= 1) v = fmaxf(v, __shfl_xor(v, o, 32));
  return v;
}

// ---- convert f32 weights -> bf16 in workspace (in_proj_w 768x256, out_w 256x256)
__global__ __launch_bounds__(256)
void convert_weights_kernel(const float* __restrict__ ipw,
                            const float* __restrict__ ow,
                            bf16* __restrict__ dst) {
  int g = blockIdx.x * 256 + threadIdx.x;
  const int n1 = 768 * 256;
  const int n2 = 256 * 256;
  if (g < n1)            dst[g] = (bf16)ipw[g];
  else if (g < n1 + n2)  dst[g] = (bf16)ow[g - n1];
}

// ---- fully fused attention layer: one workgroup per batch element
__global__ __launch_bounds__(256)
void sattn_fused_kernel(const float* __restrict__ x,
                        const float* __restrict__ qc,
                        const float* __restrict__ pe,
                        const float* __restrict__ ipb,
                        const float* __restrict__ outb,
                        const float* __restrict__ lng,
                        const float* __restrict__ lnb,
                        const bf16*  __restrict__ wqkv,   // [768][256] bf16
                        const bf16*  __restrict__ wout,   // [256][256] bf16
                        float* __restrict__ out) {
  extern __shared__ char smem[];
  bf16*     sA   = (bf16*)(smem + OFF_A);
  bf16*     sQ   = (bf16*)(smem + OFF_Q);
  bf16*     sK   = (bf16*)(smem + OFF_K);
  bf16*     sV   = (bf16*)(smem + OFF_V);   // transposed [D][KPAD]
  float*    sS   = (float*)(smem + OFF_S);
  bf16*     sP   = (bf16*)(smem + OFF_P);
  float*    sC   = (float*)(smem + OFF_C);
  unsigned* sM   = (unsigned*)(smem + OFF_M);
  float*    sRed = (float*)(smem + OFF_R);
  float*    sR   = (float*)(smem + OFF_Q);  // phase3 residual f32 [LP][D]

  const int b    = blockIdx.x;
  const int tid  = threadIdx.x;
  const int lane = tid & 31;
  const int wave = tid >> 5;
  const int l15  = lane & 15;
  const int lo16 = (lane < 16);
  const int kOffLo = lo16 ? 0 : 8;     // ISA 16-bit A/B fragment K-halves
  const int kOffHi = lo16 ? 16 : 24;
  const int rOff   = lo16 ? 0 : 8;     // C layout: VGPR r -> M=r (+8 for hi lanes)

  // ---------------- Phase 0: stage coords, h = x + pe, zero pads ----------------
  for (int i = tid; i < LP; i += 256) {
    float cx = 0.f, cy = 0.f, cz = 0.f;
    if (i < L) {
      const float* c = qc + (size_t)(b * L + i) * 3;
      cx = c[0]; cy = c[1]; cz = c[2];
    }
    sC[i*4+0] = cx; sC[i*4+1] = cy; sC[i*4+2] = cz; sC[i*4+3] = 0.f;
  }
  for (int idx = tid; idx < LP * D; idx += 256) {
    int l = idx >> 8, d = idx & 255;
    float v = 0.f;
    if (l < L) v = x[(size_t)(b * L + l) * D + d] + pe[l * D + d];
    sA[idx] = (bf16)v;
  }
  for (int idx = tid; idx < (LP - L) * KPAD; idx += 256)      // probs pad rows
    sP[(L + idx / KPAD) * KPAD + (idx % KPAD)] = (bf16)0.f;
  for (int idx = tid; idx < D * 16; idx += 256)               // V pad keys 112..127
    sV[(idx >> 4) * KPAD + 112 + (idx & 15)] = (bf16)0.f;
  if (tid == 0) sRed[0] = 0.f;
  __syncthreads();

  // ---------------- Phase 0b: coords mask (w = 0.9^cheb ; mask = w < mean) ------
  float part = 0.f;
  for (int idx = tid; idx < L * L; idx += 256) {
    int i = idx / L, j = idx - i * L;
    float dx = fabsf(sC[i*4+0] - sC[j*4+0]);
    float dy = fabsf(sC[i*4+1] - sC[j*4+1]);
    float dz = fabsf(sC[i*4+2] - sC[j*4+2]);
    part += exp2f(fmaxf(dx, fmaxf(dy, dz)) * LOG2BETA);
  }
  atomicAdd(&sRed[0], part);
  __syncthreads();
  const float meanw = sRed[0] * (1.f / (L * L));
  for (int w = tid; w < LP * 4; w += 256) {
    int i = w >> 2, jw = w & 3;
    unsigned bits = 0u;
    for (int jj = 0; jj < 32; ++jj) {
      int j = jw * 32 + jj;
      bool masked = true;
      if (i < L && j < L) {
        float dx = fabsf(sC[i*4+0] - sC[j*4+0]);
        float dy = fabsf(sC[i*4+1] - sC[j*4+1]);
        float dz = fabsf(sC[i*4+2] - sC[j*4+2]);
        masked = exp2f(fmaxf(dx, fmaxf(dy, dz)) * LOG2BETA) < meanw;
      }
      if (masked) bits |= (1u << jj);
    }
    sM[w] = bits;
  }
  __syncthreads();

  // ---------------- Phase 1: qkv = h @ W^T + b  (C: [LP][768]) ------------------
  for (int t = wave; t < 336; t += 8) {       // 7 M-tiles x 48 N-tiles
    int m0 = (t / 48) * 16;
    int n0 = (t % 48) * 16;
    int mrow = m0 + l15;
    int ncol = n0 + l15;                      // B[k][n] = W[n][k] : contiguous in k
    v8f acc = {};
#pragma unroll
    for (int kt = 0; kt < 8; ++kt) {
      int k0 = kt * 32;
      v16bf a  = frag16(&sA[mrow * D + k0 + kOffLo], &sA[mrow * D + k0 + kOffHi]);
      v16bf bm = frag16(&wqkv[ncol * D + k0 + kOffLo], &wqkv[ncol * D + k0 + kOffHi]);
      acc = wmma_bf16(a, bm, acc);
    }
    float bias = ipb[ncol];
    int mb = m0 + rOff;
    if (ncol < 256) {
#pragma unroll
      for (int r = 0; r < 8; ++r) sQ[(mb + r) * D + ncol] = (bf16)(acc[r] + bias);
    } else if (ncol < 512) {
      int cc = ncol - 256;
#pragma unroll
      for (int r = 0; r < 8; ++r) sK[(mb + r) * D + cc] = (bf16)(acc[r] + bias);
    } else {
      int cc = ncol - 512;                    // V stored transposed [feature][key]
#pragma unroll
      for (int r = 0; r < 8; ++r) sV[cc * KPAD + mb + r] = (bf16)(acc[r] + bias);
    }
  }
  __syncthreads();

  // ---------------- Phase 2: per-head attention ---------------------------------
  for (int hd = 0; hd < NHEAD; ++hd) {
    const int cbase = hd * DH;
    // scores S = Q_h K_h^T * scale, masked (K=32 -> single WMMA per tile)
    for (int t = wave; t < 49; t += 8) {
      int m0 = (t / 7) * 16, n0 = (t % 7) * 16;
      int mrow = m0 + l15;
      int nrow = n0 + l15;                    // B[k][n] = K[n][cbase+k] contiguous
      v16bf a  = frag16(&sQ[mrow * D + cbase + kOffLo], &sQ[mrow * D + cbase + kOffHi]);
      v16bf bm = frag16(&sK[nrow * D + cbase + kOffLo], &sK[nrow * D + cbase + kOffHi]);
      v8f acc = {};
      acc = wmma_bf16(a, bm, acc);
      int n  = n0 + l15;
      int mb = m0 + rOff;
#pragma unroll
      for (int r = 0; r < 8; ++r) {
        int m = mb + r;
        bool msk = (sM[m * 4 + (n >> 5)] >> (n & 31)) & 1u;
        sS[m * LP + n] = msk ? -1e30f : acc[r] * SCALE;
      }
    }
    __syncthreads();
    // softmax: one wave per row, lanes cover 4 strided columns
    for (int row = wave; row < L; row += 8) {
      float vals[4];
#pragma unroll
      for (int q2 = 0; q2 < 4; ++q2) {
        int c = lane + q2 * 32;
        vals[q2] = (c < LP) ? sS[row * LP + c] : -1e30f;
      }
      float mx = wave_max(fmaxf(fmaxf(vals[0], vals[1]), fmaxf(vals[2], vals[3])));
      float e[4]; float s = 0.f;
#pragma unroll
      for (int q2 = 0; q2 < 4; ++q2) { e[q2] = exp2f((vals[q2] - mx) * LOG2E); s += e[q2]; }
      s = wave_sum(s);
      float inv = 1.f / s;
#pragma unroll
      for (int q2 = 0; q2 < 4; ++q2) {
        int c = lane + q2 * 32;
        sP[row * KPAD + c] = (bf16)((c < L) ? e[q2] * inv : 0.f);
      }
    }
    __syncthreads();
    // O_h = P @ V_h  (B from transposed V: contiguous in k)
    for (int t = wave; t < 14; t += 8) {
      int m0   = (t >> 1) * 16;
      int ncol = cbase + (t & 1) * 16 + l15;
      int mrow = m0 + l15;
      v8f acc = {};
#pragma unroll
      for (int kt = 0; kt < 4; ++kt) {
        int k0 = kt * 32;
        v16bf a  = frag16(&sP[mrow * KPAD + k0 + kOffLo], &sP[mrow * KPAD + k0 + kOffHi]);
        v16bf bm = frag16(&sV[ncol * KPAD + k0 + kOffLo], &sV[ncol * KPAD + k0 + kOffHi]);
        acc = wmma_bf16(a, bm, acc);
      }
      int mb = m0 + rOff;
#pragma unroll
      for (int r = 0; r < 8; ++r) sA[(mb + r) * D + ncol] = (bf16)acc[r];
    }
    __syncthreads();
  }

  // ---------------- Phase 3: out proj + residual --------------------------------
  for (int t = wave; t < 112; t += 8) {       // 7 M-tiles x 16 N-tiles
    int m0 = (t >> 4) * 16;
    int n0 = (t & 15) * 16;
    int mrow = m0 + l15;
    int ncol = n0 + l15;
    v8f acc = {};
#pragma unroll
    for (int kt = 0; kt < 8; ++kt) {
      int k0 = kt * 32;
      v16bf a  = frag16(&sA[mrow * D + k0 + kOffLo], &sA[mrow * D + k0 + kOffHi]);
      v16bf bm = frag16(&wout[ncol * D + k0 + kOffLo], &wout[ncol * D + k0 + kOffHi]);
      acc = wmma_bf16(a, bm, acc);
    }
    float bias = outb[ncol];
    int mb = m0 + rOff;
#pragma unroll
    for (int r = 0; r < 8; ++r) {
      int m = mb + r;
      float h = 0.f;
      if (m < L) h = x[(size_t)(b * L + m) * D + ncol] + pe[m * D + ncol];
      sR[m * D + ncol] = acc[r] + bias + h;
    }
  }
  __syncthreads();

  // ---------------- Phase 4: layernorm (wave per row) ---------------------------
  for (int row = wave; row < L; row += 8) {
    float vals[8]; float s = 0.f;
#pragma unroll
    for (int q2 = 0; q2 < 8; ++q2) {
      int d = lane + q2 * 32;
      vals[q2] = sR[row * D + d];
      s += vals[q2];
    }
    s = wave_sum(s);
    float mu = s * (1.f / D);
    float v2 = 0.f;
#pragma unroll
    for (int q2 = 0; q2 < 8; ++q2) { float dv = vals[q2] - mu; v2 += dv * dv; }
    v2 = wave_sum(v2);
    float rstd = rsqrtf(v2 * (1.f / D) + LN_EPS);
    float* o = out + (size_t)(b * L + row) * D;
#pragma unroll
    for (int q2 = 0; q2 < 8; ++q2) {
      int d = lane + q2 * 32;
      o[d] = (vals[q2] - mu) * rstd * lng[d] + lnb[d];
    }
  }
}

extern "C" void kernel_launch(void* const* d_in, const int* in_sizes, int n_in,
                              void* d_out, int out_size, void* d_ws, size_t ws_size,
                              hipStream_t stream) {
  (void)in_sizes; (void)n_in; (void)out_size; (void)ws_size;
  const float* x   = (const float*)d_in[0];
  const float* qc  = (const float*)d_in[1];
  const float* pe  = (const float*)d_in[2];
  const float* ipw = (const float*)d_in[3];
  const float* ipb = (const float*)d_in[4];
  const float* ow  = (const float*)d_in[5];
  const float* ob  = (const float*)d_in[6];
  const float* lng = (const float*)d_in[7];
  const float* lnb = (const float*)d_in[8];
  bf16* wb = (bf16*)d_ws;   // [768*256] qkv weights + [256*256] out weights, bf16

  (void)hipFuncSetAttribute(reinterpret_cast<const void*>(sattn_fused_kernel),
                            hipFuncAttributeMaxDynamicSharedMemorySize, SMEM_BYTES);

  convert_weights_kernel<<<1024, 256, 0, stream>>>(ipw, ow, wb);
  sattn_fused_kernel<<<BATCH, 256, SMEM_BYTES, stream>>>(
      x, qc, pe, ipb, ob, lng, lnb, wb, wb + 768 * 256, (float*)d_out);
}
